// RSA_84009560310348
// MI455X (gfx1250) — compile-verified, matrix operand order checked by reading, not run on hardware
//
#include <hip/hip_runtime.h>

// ---------------------------------------------------------------------------
// MI455X (gfx1250, wave32) RSA block.
//   cvt  : x, w_in, w_out  f32 -> bf16 (bandwidth-bound, one pass)
//   GEMM1: h = x @ w_in^T + b_in   (2048x3584x2048)  bf16 WMMA, async-LDS DB
//   conv : depthwise causal K=4 + silu, split q/k/v/f, k-norm, g=sigmoid(f')
//   scan : 16 WGs (B*H); St@W via v_wmma_f32_16x16x32_bf16; f32 state in regs,
//          bf16 mirror in LDS; async double-buffered per-step input staging
//   norm : (y + v*D) * silu(gate), RMSNorm * g_w  -> bf16
//   GEMM2: out = yn @ w_out^T      (2048x2048x1024)  bf16 WMMA, async-LDS DB
// ---------------------------------------------------------------------------

typedef __attribute__((ext_vector_type(16))) __bf16 v16bf;
typedef __attribute__((ext_vector_type(8)))  float  v8f;
typedef __attribute__((ext_vector_type(4)))  int    i32x4_t;

#define S_    1024
#define H_    8
#define DK_   64
#define DV_   128
#define CONVD 2560
#define NHT   3584   // CONVD + G
#define G_    1024

// ---- CDNA5 async global->LDS path (guarded; falls back to direct copy) ----
#if defined(__has_builtin)
#  if __has_builtin(__builtin_amdgcn_global_load_async_to_lds_b128)
#    define HAVE_ASYNC_LDS 1
#  else
#    define HAVE_ASYNC_LDS 0
#  endif
#  if __has_builtin(__builtin_amdgcn_s_wait_asynccnt)
#    define WAIT_ASYNC(n) __builtin_amdgcn_s_wait_asynccnt(n)
#  else
#    define WAIT_ASYNC(n) asm volatile("s_wait_asynccnt %0" :: "i"(n) : "memory")
#  endif
#else
#  define HAVE_ASYNC_LDS 0
#  define WAIT_ASYNC(n) asm volatile("s_wait_asynccnt %0" :: "i"(n) : "memory")
#endif

static __device__ __forceinline__ void async_cp16(const void* g, void* l) {
#if HAVE_ASYNC_LDS
  __builtin_amdgcn_global_load_async_to_lds_b128(
      (__attribute__((address_space(1))) i32x4_t*)g,
      (__attribute__((address_space(3))) i32x4_t*)l, 0, 0);
#else
  *(uint4*)l = *(const uint4*)g;
#endif
}

static __device__ __forceinline__ unsigned short f2bf(float f) {
  unsigned int u = __float_as_uint(f);
  u += 0x7fffu + ((u >> 16) & 1u);          // round-to-nearest-even
  return (unsigned short)(u >> 16);
}

struct U16x { unsigned short u[16]; };

// A/B fragment for V_WMMA_*_16X16X32_BF16 (wave32):
//  lanes 0-15 : row = lane,    K = {0..7, 16..23}
//  lanes16-31 : row = lane-16, K = {8..15, 24..31}
// p = &tile[row*stride + 8*half (+32*ks)]
static __device__ __forceinline__ v16bf frag_u16(const unsigned short* p) {
  U16x t;
#pragma unroll
  for (int g = 0; g < 4; ++g) {
    const int o = (g & 1) * 4 + (g >> 1) * 16;   // {0,4,16,20}
#pragma unroll
    for (int e = 0; e < 4; ++e) t.u[g*4 + e] = p[o + e];
  }
  return __builtin_bit_cast(v16bf, t);
}

static __device__ __forceinline__ v8f vzero8() {
  v8f z;
#pragma unroll
  for (int i = 0; i < 8; ++i) z[i] = 0.f;
  return z;
}

// ---------------------------------------------------------------------------
// f32 -> bf16 pack, 8 elems/thread (all sizes are multiples of 2048).
// ---------------------------------------------------------------------------
__global__ __launch_bounds__(256) void cvt_bf16(
    const float* __restrict__ s, unsigned short* __restrict__ d)
{
  const size_t i = ((size_t)blockIdx.x * 256 + threadIdx.x) * 8;
  const float4 a = *(const float4*)(s + i);
  const float4 b = *(const float4*)(s + i + 4);
  unsigned short t[8] = { f2bf(a.x), f2bf(a.y), f2bf(a.z), f2bf(a.w),
                          f2bf(b.x), f2bf(b.y), f2bf(b.z), f2bf(b.w) };
  *(uint4*)(d + i) = *(const uint4*)t;
}

// ---------------------------------------------------------------------------
// C[M,N] = A[M,K] @ Bw[N,K]^T (+bias).  A,Bw bf16, C f32.
// 128x128 tile, K-step 32, double-buffered async LDS staging.
// 8 waves: each wave computes 32(M) x 64(N) = 8 WMMAs per K-step.
// ---------------------------------------------------------------------------
#define GT_LDS 40   // ushort row stride (80B: keeps 16B alignment for chunks)

__global__ __launch_bounds__(256) void gemm_bf16_nt(
    const unsigned short* __restrict__ A, const unsigned short* __restrict__ Bw,
    const float* __restrict__ bias, float* __restrict__ C,
    int M, int N, int Kd)
{
  __shared__ unsigned short smA[2][128 * GT_LDS];
  __shared__ unsigned short smB[2][128 * GT_LDS];

  const int tid  = threadIdx.x;
  const int wave = tid >> 5, lane = tid & 31;
  const int half = lane >> 4, l15 = lane & 15;
  const int m0 = blockIdx.y * 128, n0 = blockIdx.x * 128;
  const int wm = (wave & 3) * 32;
  const int wn = (wave >> 2) * 64;

  // staging role: 2 threads per row, 16 bf16 (2 x b128) each
  const int srow = tid >> 1;
  const int sc0  = (tid & 1) * 16;

  auto stage = [&](int buf, int k0) {
#pragma unroll
    for (int c = 0; c < 2; ++c) {
      const int cc = sc0 + c * 8;
      async_cp16(A + (size_t)(m0 + srow) * Kd + k0 + cc,
                 &smA[buf][srow * GT_LDS + cc]);
      async_cp16(Bw + (size_t)(n0 + srow) * Kd + k0 + cc,
                 &smB[buf][srow * GT_LDS + cc]);
    }
  };

  v8f acc[2][4];
#pragma unroll
  for (int i = 0; i < 2; ++i)
#pragma unroll
    for (int j = 0; j < 4; ++j) acc[i][j] = vzero8();

  stage(0, 0);
  for (int k0 = 0; k0 < Kd; k0 += 32) {
    const int cur = (k0 >> 5) & 1;
    if (k0 + 32 < Kd) { stage(cur ^ 1, k0 + 32); WAIT_ASYNC(4); }
    else              { WAIT_ASYNC(0); }
    __syncthreads();

    const v16bf af0 = frag_u16(&smA[cur][(wm      + l15) * GT_LDS + half * 8]);
    const v16bf af1 = frag_u16(&smA[cur][(wm + 16 + l15) * GT_LDS + half * 8]);
#pragma unroll
    for (int nt = 0; nt < 4; ++nt) {
      const v16bf bf = frag_u16(&smB[cur][(wn + nt * 16 + l15) * GT_LDS + half * 8]);
      acc[0][nt] = __builtin_amdgcn_wmma_f32_16x16x32_bf16(
          false, af0, false, bf, (short)0, acc[0][nt], false, false);
      acc[1][nt] = __builtin_amdgcn_wmma_f32_16x16x32_bf16(
          false, af1, false, bf, (short)0, acc[1][nt], false, false);
    }
    __syncthreads();
  }

#pragma unroll
  for (int i = 0; i < 2; ++i)
#pragma unroll
    for (int nt = 0; nt < 4; ++nt) {
      const int cc = n0 + wn + nt * 16 + l15;
      const float bv = bias ? bias[cc] : 0.f;
      const int rb = m0 + wm + i * 16 + half * 8;
#pragma unroll
      for (int j = 0; j < 8; ++j)
        C[(size_t)(rb + j) * N + cc] = acc[i][nt][j] + bv;
    }
}

// ---------------------------------------------------------------------------
// Depthwise causal conv (K=4) + silu + split/normalize.  One block per token.
// ---------------------------------------------------------------------------
__global__ __launch_bounds__(256) void conv_split(
    const float* __restrict__ hbuf, const float* __restrict__ cw,
    const float* __restrict__ cb,   const float* __restrict__ fmul,
    const float* __restrict__ fbias,
    float* __restrict__ qb, float* __restrict__ kb,
    float* __restrict__ gb, float* __restrict__ vb)
{
  __shared__ float yc[CONVD];
  __shared__ float knorm[H_];
  const int bs = blockIdx.x;
  const int b = bs >> 10, s = bs & 1023;
  const int tid = threadIdx.x;
  if (tid < H_) knorm[tid] = 0.f;

  for (int c = tid; c < CONVD; c += 256) {
    float acc = cb[c];
#pragma unroll
    for (int j = 0; j < 4; ++j) {
      const int ss = s - 3 + j;
      const float xv = (ss >= 0)
          ? hbuf[((size_t)(b * 1024 + ss)) * NHT + c] : 0.f;
      acc = fmaf(cw[c * 4 + j], xv, acc);
    }
    yc[c] = acc / (1.f + __expf(-acc));              // silu
  }
  __syncthreads();

  for (int c = tid; c < 512; c += 256) {
    const float kv = yc[512 + c];
    atomicAdd(&knorm[c >> 6], kv * kv);
  }
  __syncthreads();

  for (int c = tid; c < 512; c += 256) {
    const int hh = c >> 6, dk = c & 63;
    const size_t o = (((size_t)(b * H_ + hh)) * S_ + s) * DK_ + dk;
    qb[o] = yc[c];
    kb[o] = yc[512 + c] * rsqrtf(knorm[hh] + 1e-12f);
    const float fm = 2.f / (1.f + __expf(-fmul[hh]));
    const float ft = fm * (yc[2048 + c] + fbias[hh * 64 + dk]);
    gb[o] = 1.f / (1.f + __expf(-ft));               // sigmoid(f')
  }
  for (int c = tid; c < 1024; c += 256) {
    const int hh = c >> 7, dv = c & 127;
    vb[(((size_t)(b * H_ + hh)) * S_ + s) * DV_ + dv] = yc[1024 + c];
  }
}

// ---------------------------------------------------------------------------
// Recurrent scan.  grid = B*H = 16 blocks, 8 waves.
// f32 state in registers (C-layout ownership); bf16 mirror in LDS feeds the
// A fragments of cand_pre = St(64x128) @ W(128x128); per-step inputs are
// double-buffered via async global->LDS loads (prefetch t+1 behind step t).
// ---------------------------------------------------------------------------
#define WST_ 132

__global__ __launch_bounds__(256) void rsa_scan(
    const float* __restrict__ qb, const float* __restrict__ kb,
    const float* __restrict__ gb, const float* __restrict__ vb,
    const float* __restrict__ W,  float* __restrict__ yb)
{
  __shared__ unsigned short stS [DK_ * WST_];   // bf16 mirror of St
  __shared__ unsigned short stWt[DV_ * WST_];   // W^T as bf16: [w][v]
  __shared__ float          stIn[2][3 * DK_ + DV_];   // q|k|g|v double buffer
  __shared__ float          stY [DV_];

  const int bh = blockIdx.x;
  const int h  = bh & (H_ - 1);
  const int tid  = threadIdx.x;
  const int wave = tid >> 5, lane = tid & 31;
  const int half = lane >> 4, l15 = lane & 15;
  const int mt = wave & 3, nh = wave >> 2;

  const float* qp = qb + (size_t)bh * S_ * DK_;
  const float* kp = kb + (size_t)bh * S_ * DK_;
  const float* gp = gb + (size_t)bh * S_ * DK_;
  const float* vp = vb + (size_t)bh * S_ * DV_;
  float*       yp = yb + (size_t)bh * S_ * DV_;

  // async staging: tids 0..79 each move one 16B chunk (q:16, k:16, g:16, v:32)
  auto issue_in = [&](int buf, int t) {
    if (tid < 80) {
      const float* gsrc; int loff;
      if (tid < 16)      { gsrc = qp + (size_t)t * DK_ + (tid      ) * 4; loff = (tid      ) * 4; }
      else if (tid < 32) { gsrc = kp + (size_t)t * DK_ + (tid - 16 ) * 4; loff =  64 + (tid - 16) * 4; }
      else if (tid < 48) { gsrc = gp + (size_t)t * DK_ + (tid - 32 ) * 4; loff = 128 + (tid - 32) * 4; }
      else               { gsrc = vp + (size_t)t * DV_ + (tid - 48 ) * 4; loff = 192 + (tid - 48) * 4; }
      async_cp16(gsrc, &stIn[buf][loff]);
    }
  };

  const float* Wh = W + (size_t)h * DV_ * DV_;
  for (int i = tid; i < DV_ * DV_; i += 256) {   // transpose W -> [w][v]
    const int vv = i >> 7, ww = i & 127;
    stWt[ww * WST_ + vv] = f2bf(Wh[i]);
  }
  for (int i = tid; i < DK_ * WST_; i += 256) stS[i] = 0;
  if (tid < DV_) stY[tid] = 0.f;

  float Sreg[4][8];
#pragma unroll
  for (int nt = 0; nt < 4; ++nt)
#pragma unroll
    for (int j = 0; j < 8; ++j) Sreg[nt][j] = 0.f;

  issue_in(0, 0);
  __syncthreads();

  for (int t = 0; t < S_; ++t) {
    // prefetch t+1 behind this step's compute, then wait for t's data
    if (t + 1 < S_) { issue_in((t + 1) & 1, t + 1); WAIT_ASYNC(1); }
    else            { WAIT_ASYNC(0); }
    __syncthreads();
    const float* inq = stIn[t & 1];
    const float* ink = inq + DK_;
    const float* ing = inq + 2 * DK_;
    const float* inv = inq + 3 * DK_;

    // ---- cand_pre = St @ W  (16 WMMAs per wave) ----
    v16bf af[4];
#pragma unroll
    for (int ks = 0; ks < 4; ++ks)
      af[ks] = frag_u16(&stS[(mt * 16 + l15) * WST_ + ks * 32 + half * 8]);

    v8f acc[4];
#pragma unroll
    for (int nt = 0; nt < 4; ++nt) acc[nt] = vzero8();
#pragma unroll
    for (int nt = 0; nt < 4; ++nt)
#pragma unroll
      for (int ks = 0; ks < 4; ++ks) {
        const v16bf bf =
            frag_u16(&stWt[(nh * 64 + nt * 16 + l15) * WST_ + ks * 32 + half * 8]);
        acc[nt] = __builtin_amdgcn_wmma_f32_16x16x32_bf16(
            false, af[ks], false, bf, (short)0, acc[nt], false, false);
      }

    float cand[4][8];
#pragma unroll
    for (int nt = 0; nt < 4; ++nt) {
      const float vcol = inv[nh * 64 + nt * 16 + l15];
#pragma unroll
      for (int j = 0; j < 8; ++j)
        cand[nt][j] = tanhf(vcol + acc[nt][j]);
    }
    __syncthreads();   // all waves done reading stS fragments

    // ---- St = g*St + k*cand (f32 regs); y partials; publish bf16 mirror ----
    float ysum[4] = {0.f, 0.f, 0.f, 0.f};
#pragma unroll
    for (int j = 0; j < 8; ++j) {
      const int row = mt * 16 + half * 8 + j;
      const float gr = ing[row], kr = ink[row], qr = inq[row];
#pragma unroll
      for (int nt = 0; nt < 4; ++nt) {
        const int col = nh * 64 + nt * 16 + l15;
        const float snew = fmaf(gr, Sreg[nt][j], kr * cand[nt][j]);
        Sreg[nt][j] = snew;
        ysum[nt] = fmaf(qr, snew, ysum[nt]);
        stS[row * WST_ + col] = f2bf(snew);
      }
    }

    // lane L <-> L^16 share a column: SWAPX16 then 4-way LDS atomic per col
#pragma unroll
    for (int nt = 0; nt < 4; ++nt) {
      const int o = __builtin_amdgcn_ds_swizzle(__float_as_int(ysum[nt]), 0x401f);
      ysum[nt] += __int_as_float(o);
    }
    if (half == 0) {
#pragma unroll
      for (int nt = 0; nt < 4; ++nt)
        atomicAdd(&stY[nh * 64 + nt * 16 + l15], ysum[nt]);
    }
    __syncthreads();
    if (tid < DV_) { yp[(size_t)t * DV_ + tid] = stY[tid]; stY[tid] = 0.f; }
    __syncthreads();
  }
}

// ---------------------------------------------------------------------------
// (y + v*D) * silu(gate), RMSNorm over G=1024, * g_w -> bf16 for GEMM2.
// ---------------------------------------------------------------------------
__global__ __launch_bounds__(256) void gate_norm(
    const float* __restrict__ hbuf, const float* __restrict__ yb,
    const float* __restrict__ vb,   const float* __restrict__ Dp,
    const float* __restrict__ gw,   unsigned short* __restrict__ yn)
{
  __shared__ float red[8];
  const int bs = blockIdx.x;
  const int b = bs >> 10, s = bs & 1023;
  const int tid = threadIdx.x;

  float t[4]; float ss = 0.f;
#pragma unroll
  for (int r = 0; r < 4; ++r) {
    const int g = tid + 256 * r;
    const int hh = g >> 7, dv = g & 127;
    const size_t o = (((size_t)(b * H_ + hh)) * S_ + s) * DV_ + dv;
    const float yv   = yb[o] + vb[o] * Dp[g];
    const float gate = hbuf[(size_t)bs * NHT + CONVD + g];
    const float val  = yv * (gate / (1.f + __expf(-gate)));
    t[r] = val; ss = fmaf(val, val, ss);
  }
#pragma unroll
  for (int off = 16; off; off >>= 1) ss += __shfl_xor(ss, off, 32);
  if ((tid & 31) == 0) red[tid >> 5] = ss;
  __syncthreads();
  float tot = 0.f;
#pragma unroll
  for (int i = 0; i < 8; ++i) tot += red[i];
  const float inv = rsqrtf(tot * (1.f / 1024.f) + 1e-6f);
#pragma unroll
  for (int r = 0; r < 4; ++r) {
    const int g = tid + 256 * r;
    yn[(size_t)bs * G_ + g] = f2bf(t[r] * inv * gw[g]);
  }
}

// ---------------------------------------------------------------------------
extern "C" void kernel_launch(void* const* d_in, const int* in_sizes, int n_in,
                              void* d_out, int out_size, void* d_ws, size_t ws_size,
                              hipStream_t stream)
{
  (void)in_sizes; (void)n_in; (void)out_size; (void)ws_size;
  const float* x      = (const float*)d_in[0];
  const float* w_in   = (const float*)d_in[1];
  const float* b_in   = (const float*)d_in[2];
  const float* conv_w = (const float*)d_in[3];
  const float* conv_b = (const float*)d_in[4];
  const float* Dp     = (const float*)d_in[5];
  const float* Wst    = (const float*)d_in[6];
  const float* fmul   = (const float*)d_in[7];
  const float* fbias  = (const float*)d_in[8];
  const float* gw     = (const float*)d_in[9];
  const float* w_out  = (const float*)d_in[10];
  float* out = (float*)d_out;

  // f32 region
  float* hbuf = (float*)d_ws;                       // 2048*3584
  float* qbuf = hbuf + (size_t)2048 * 3584;         // 16*1024*64
  float* kbuf = qbuf + (size_t)16 * 1024 * 64;
  float* gbuf = kbuf + (size_t)16 * 1024 * 64;
  float* vbuf = gbuf + (size_t)16 * 1024 * 64;      // 16*1024*128
  float* ybuf = vbuf + (size_t)16 * 1024 * 128;     // 16*1024*128
  // bf16 region
  unsigned short* xb  = (unsigned short*)(ybuf + (size_t)16 * 1024 * 128);
  unsigned short* wib = xb  + (size_t)2048 * 2048;  // 3584*2048
  unsigned short* wob = wib + (size_t)3584 * 2048;  // 2048*1024
  unsigned short* ynb = wob + (size_t)2048 * 1024;  // 2048*1024

  cvt_bf16<<<(2048 * 2048) / 2048, 256, 0, stream>>>(x, xb);
  cvt_bf16<<<(3584 * 2048) / 2048, 256, 0, stream>>>(w_in, wib);
  cvt_bf16<<<(2048 * 1024) / 2048, 256, 0, stream>>>(w_out, wob);

  gemm_bf16_nt<<<dim3(NHT / 128, 2048 / 128), 256, 0, stream>>>(
      xb, wib, b_in, hbuf, 2048, NHT, 2048);
  conv_split<<<2048, 256, 0, stream>>>(
      hbuf, conv_w, conv_b, fmul, fbias, qbuf, kbuf, gbuf, vbuf);
  rsa_scan<<<16, 256, 0, stream>>>(qbuf, kbuf, gbuf, vbuf, Wst, ybuf);
  gate_norm<<<2048, 256, 0, stream>>>(hbuf, ybuf, vbuf, Dp, gw, ynb);
  gemm_bf16_nt<<<dim3(2048 / 128, 2048 / 128), 256, 0, stream>>>(
      ynb, wob, nullptr, out, 2048, 2048, 1024);
}
